// NeighborCooccurrenceEncoder_73701638799825
// MI455X (gfx1250) — matrix operand order repaired and checked by reading, not compiled, not run on hardware
//
#include <hip/hip_runtime.h>
#include <hip/hip_bf16.h>

typedef __attribute__((ext_vector_type(16))) _Float16 v16h;
typedef __attribute__((ext_vector_type(8)))  float    v8f;

#define BATCH 1024
#define SEQL  256
#define DIM   64
#define TROWS 257   // counts 0..256

// ---------------------------------------------------------------------------
// Kernel 1: build the 257x64 MLP lookup table with WMMA.
//   table[a][e] = b2[e] + sum_d relu(a*W1[d] + b1[d]) * W2[e][d]
// H is (272 x 64) f32, split into hi/lo f16 halves; W2^T likewise. Three
// v_wmma_f32_16x16x32_f16 per K-step give ~f32 accuracy in the f32 accum.
// Launch: 1 block x 544 threads = 17 waves; wave w owns row-tile [16w,16w+16).
// ---------------------------------------------------------------------------
__global__ __launch_bounds__(544) void build_table_kernel(
    const float* __restrict__ W1, const float* __restrict__ b1,
    const float* __restrict__ W2, const float* __restrict__ b2,
    float* __restrict__ table)
{
    const int lane = threadIdx.x & 31;
    const int wave = threadIdx.x >> 5;          // 0..16 -> M tile
    const int half = lane >> 4;                 // lane group 0/1
    const int rc   = lane & 15;                 // A row / B,D column within tile
    const float a  = (float)(wave * 16 + rc);   // count value for this A row

    for (int nt = 0; nt < 4; ++nt) {            // N tiles: columns e
        const int ncol = nt * 16 + rc;          // global output column (B/D col)
        v8f c;
        const float bias = b2[ncol];
        #pragma unroll
        for (int m = 0; m < 8; ++m) c[m] = bias;   // fold b2 into accumulator

        #pragma unroll
        for (int ks = 0; ks < 2; ++ks) {        // K = ks*32 .. ks*32+31
            v16h ahi, alo, bhi, blo;
            #pragma unroll
            for (int i = 0; i < 8; ++i) {       // VGPR index of v16h
                #pragma unroll
                for (int j = 0; j < 2; ++j) {   // low/high half-word
                    // A layout (16-bit 16x32): VGPR0-3 -> K=2i+j+8*half,
                    //                          VGPR4-7 -> K=16+2(i-4)+j+8*half
                    const int ka = ks * 32 + ((i >> 2) * 16) + ((i & 3) * 2) + j + 8 * half;
                    const float h = fmaxf(a * W1[ka] + b1[ka], 0.0f);
                    const _Float16 hh = (_Float16)h;
                    ahi[2 * i + j] = hh;
                    alo[2 * i + j] = (_Float16)(h - (float)hh);
                    // B layout (16-bit 32x16): same K order as A
                    const int kb = ks * 32 + ((i >> 2) * 16) + ((i & 3) * 2) + j + 8 * half;
                    const float w = W2[ncol * DIM + kb];   // W2^T[kb][ncol]
                    const _Float16 wh = (_Float16)w;
                    bhi[2 * i + j] = wh;
                    blo[2 * i + j] = (_Float16)(w - (float)wh);
                }
            }
            c = __builtin_amdgcn_wmma_f32_16x16x32_f16(false, ahi, false, bhi, (short)0, c, false, false);
            c = __builtin_amdgcn_wmma_f32_16x16x32_f16(false, ahi, false, blo, (short)0, c, false, false);
            c = __builtin_amdgcn_wmma_f32_16x16x32_f16(false, alo, false, bhi, (short)0, c, false, false);
        }
        // D layout: VGPR m -> row m + 8*half, col = rc
        #pragma unroll
        for (int m = 0; m < 8; ++m) {
            const int row = wave * 16 + m + 8 * half;
            if (row < TROWS) table[row * DIM + nt * 16 + rc] = c[m];
        }
    }
}

// ---------------------------------------------------------------------------
// Kernel 2: per-batch co-occurrence counts + table gather + coalesced store.
// One block (256 threads) per batch. Output-bandwidth bound (134 MB writes).
// ---------------------------------------------------------------------------
__global__ __launch_bounds__(256) void cooc_gather_kernel(
    const int* __restrict__ src_ids, const int* __restrict__ dst_ids,
    const float* __restrict__ table,
    float* __restrict__ src_feat, float* __restrict__ dst_feat)
{
    __shared__ int s_src[SEQL];
    __shared__ int s_dst[SEQL];
    __shared__ int s_css[SEQL], s_csd[SEQL], s_cds[SEQL], s_cdd[SEQL];

    const int b = blockIdx.x;
    const int t = threadIdx.x;

    s_src[t] = src_ids[b * SEQL + t];
    s_dst[t] = dst_ids[b * SEQL + t];
    // Warm the table toward the caches (emits global_prefetch_b8); 256 prefetches,
    // one per 256 B row, cover rows 0..255 of the table across the 256 threads.
    __builtin_prefetch(table + t * DIM, 0, 3);
    __syncthreads();

    const int ms = s_src[t];
    const int md = s_dst[t];
    int css = 0, csd = 0, cds = 0, cdd = 0;
    #pragma unroll 8
    for (int j = 0; j < SEQL; ++j) {
        const int sj = s_src[j];   // LDS broadcast reads
        const int dj = s_dst[j];
        css += (ms == sj);
        csd += (ms == dj);
        cds += (md == sj);
        cdd += (md == dj);
    }
    if (ms == 0) { css = 0; csd = 0; }
    if (md == 0) { cds = 0; cdd = 0; }
    s_css[t] = css; s_csd[t] = csd; s_cds[t] = cds; s_cdd[t] = cdd;
    __syncthreads();

    // Coalesced write: threads 0..63 emit row 0 cols 0..63, etc.
    const size_t base = (size_t)b * SEQL * DIM;
    for (int idx = t; idx < SEQL * DIM; idx += 256) {
        const int pos = idx >> 6;
        const int e   = idx & 63;
        src_feat[base + idx] = table[s_css[pos] * DIM + e] + table[s_csd[pos] * DIM + e];
        dst_feat[base + idx] = table[s_cds[pos] * DIM + e] + table[s_cdd[pos] * DIM + e];
    }
}

extern "C" void kernel_launch(void* const* d_in, const int* in_sizes, int n_in,
                              void* d_out, int out_size, void* d_ws, size_t ws_size,
                              hipStream_t stream) {
    const int*   src_ids = (const int*)d_in[0];     // [B, 256]
    const int*   dst_ids = (const int*)d_in[1];     // [B, 256]
    const float* W1      = (const float*)d_in[2];   // [64, 1]
    const float* b1      = (const float*)d_in[3];   // [64]
    const float* W2      = (const float*)d_in[4];   // [64, 64]
    const float* b2      = (const float*)d_in[5];   // [64]

    float* table    = (float*)d_ws;                 // 257*64*4 = 65,792 B
    float* src_feat = (float*)d_out;                // [B, 256, 64]
    float* dst_feat = src_feat + (size_t)BATCH * SEQL * DIM;

    build_table_kernel<<<1, 544, 0, stream>>>(W1, b1, W2, b2, table);
    cooc_gather_kernel<<<BATCH, 256, 0, stream>>>(src_ids, dst_ids, table,
                                                  src_feat, dst_feat);
}
